// Whitening2dIterNorm_59648505807603
// MI455X (gfx1250) — matrix operand order repaired and checked by reading, not compiled, not run on hardware
//
#include <hip/hip_runtime.h>

// ---------------------------------------------------------------------------
// IterNorm whitening for MI455X (gfx1250, wave32, WMMA bf16 path)
// Round 2: register double-buffering + wider covariance tile (32x64/wave)
// ---------------------------------------------------------------------------

typedef __bf16 bf16;
typedef __attribute__((ext_vector_type(16))) __bf16 v16bf;
typedef __attribute__((ext_vector_type(8)))  float  v8f;

#define N_ROWS 65536
#define C_DIM  512
#define SPLITS 32
#define KCHUNK (N_ROWS / SPLITS)   // 2048

// ---- WMMA fragment helpers (ISA 7.12.2 wave32 layouts) ---------------------
__device__ __forceinline__ v16bf load_frag_a(const bf16* __restrict__ base, long ld,
                                             int row0, int k0, int lane) {
    const int m  = lane & 15;
    const int kb = (lane >> 4) << 3;
    const bf16* p = base + (long)(row0 + m) * ld + (k0 + kb);
    v16bf a;
    *((uint4*)&a + 0) = *((const uint4*)p);          // K = kb .. kb+7
    *((uint4*)&a + 1) = *((const uint4*)(p + 16));   // K = kb+16 .. kb+23
    return a;
}

__device__ __forceinline__ v16bf load_frag_bt(const bf16* __restrict__ bt, long ld,
                                              int n0, int k0, int lane) {
    const int n  = lane & 15;
    const int kb = (lane >> 4) << 3;
    const bf16* p = bt + (long)(n0 + n) * ld + (k0 + kb);
    v16bf b;
    *((uint4*)&b + 0) = *((const uint4*)p + 0);
    *((uint4*)&b + 1) = *((const uint4*)p + 2);
    return b;
}

__device__ __forceinline__ v8f wmma_bf16(v16bf a, v16bf b, v8f c) {
    return __builtin_amdgcn_wmma_f32_16x16x32_bf16(false, a, false, b, (short)0, c,
                                                   false, false);
}

// ---- 1) column mean, deterministic two-stage -------------------------------
__global__ void __launch_bounds__(512) k_mean_partial(const float* __restrict__ x,
                                                      float* __restrict__ part) {
    const int c = threadIdx.x;           // 512 columns
    const int b = blockIdx.x;            // 256 blocks x 256 rows
    float s = 0.f;
    #pragma unroll 4
    for (int r = 0; r < 256; ++r)
        s += x[(size_t)(b * 256 + r) * C_DIM + c];
    part[(size_t)b * C_DIM + c] = s;
}

__global__ void __launch_bounds__(512) k_mean_final(const float* __restrict__ part,
                                                    float* __restrict__ mean) {
    const int c = threadIdx.x;
    float s = 0.f;
    #pragma unroll 4
    for (int b = 0; b < 256; ++b)
        s += part[(size_t)b * C_DIM + c];
    mean[c] = s * (1.0f / (float)N_ROWS);
}

// ---- 2) center, convert to bf16, and emit transposed copy ------------------
__global__ void __launch_bounds__(256) k_center_transpose(const float* __restrict__ x,
                                                          const float* __restrict__ mean,
                                                          bf16* __restrict__ xnbf,
                                                          bf16* __restrict__ xnT) {
    __shared__ bf16 tile[64][72];            // padded to dodge bank conflicts
    const int n0 = blockIdx.x * 64;
    const int c0 = blockIdx.y * 64;
    const int t  = threadIdx.x;
    #pragma unroll
    for (int i = 0; i < 16; ++i) {
        int idx = t + i * 256;
        int r = idx >> 6, c = idx & 63;
        float v = x[(size_t)(n0 + r) * C_DIM + c0 + c] - mean[c0 + c];
        bf16 h = (bf16)v;
        xnbf[(size_t)(n0 + r) * C_DIM + c0 + c] = h;
        tile[c][r] = h;
    }
    __syncthreads();
    #pragma unroll
    for (int i = 0; i < 16; ++i) {
        int idx = t + i * 256;
        int cc = idx >> 6, nn = idx & 63;
        xnT[(size_t)(c0 + cc) * N_ROWS + n0 + nn] = tile[cc][nn];
    }
}

// ---- 3) covariance partials: S = Xn^T Xn, K split 32 ways ------------------
// one wave per block; wave computes a 32(M) x 64(N) tile over a 2048-row chunk
// with register double-buffering: next k-step's fragments load while the
// current step's 8 WMMAs issue.
__global__ void __launch_bounds__(32) k_cov_partial(const bf16* __restrict__ xnT,
                                                    float* __restrict__ covp) {
    const int lane = threadIdx.x;
    const int i0 = blockIdx.x * 32;
    const int j0 = blockIdx.y * 64;
    const int z  = blockIdx.z;
    const int kbeg = z * KCHUNK;
    const int kend = kbeg + KCHUNK;

    v8f acc[8] = {};

    v16bf a0 = load_frag_a(xnT, N_ROWS, i0,      kbeg, lane);
    v16bf a1 = load_frag_a(xnT, N_ROWS, i0 + 16, kbeg, lane);
    v16bf b0 = load_frag_bt(xnT, N_ROWS, j0,      kbeg, lane);
    v16bf b1 = load_frag_bt(xnT, N_ROWS, j0 + 16, kbeg, lane);
    v16bf b2 = load_frag_bt(xnT, N_ROWS, j0 + 32, kbeg, lane);
    v16bf b3 = load_frag_bt(xnT, N_ROWS, j0 + 48, kbeg, lane);

    for (int k = kbeg + 32; k < kend; k += 32) {
        // prefetch next k-step
        v16bf na0 = load_frag_a(xnT, N_ROWS, i0,      k, lane);
        v16bf na1 = load_frag_a(xnT, N_ROWS, i0 + 16, k, lane);
        v16bf nb0 = load_frag_bt(xnT, N_ROWS, j0,      k, lane);
        v16bf nb1 = load_frag_bt(xnT, N_ROWS, j0 + 16, k, lane);
        v16bf nb2 = load_frag_bt(xnT, N_ROWS, j0 + 32, k, lane);
        v16bf nb3 = load_frag_bt(xnT, N_ROWS, j0 + 48, k, lane);
        // consume current fragments (loads above fly in parallel)
        acc[0] = wmma_bf16(a0, b0, acc[0]);
        acc[1] = wmma_bf16(a0, b1, acc[1]);
        acc[2] = wmma_bf16(a0, b2, acc[2]);
        acc[3] = wmma_bf16(a0, b3, acc[3]);
        acc[4] = wmma_bf16(a1, b0, acc[4]);
        acc[5] = wmma_bf16(a1, b1, acc[5]);
        acc[6] = wmma_bf16(a1, b2, acc[6]);
        acc[7] = wmma_bf16(a1, b3, acc[7]);
        a0 = na0; a1 = na1; b0 = nb0; b1 = nb1; b2 = nb2; b3 = nb3;
    }
    // tail step
    acc[0] = wmma_bf16(a0, b0, acc[0]);
    acc[1] = wmma_bf16(a0, b1, acc[1]);
    acc[2] = wmma_bf16(a0, b2, acc[2]);
    acc[3] = wmma_bf16(a0, b3, acc[3]);
    acc[4] = wmma_bf16(a1, b0, acc[4]);
    acc[5] = wmma_bf16(a1, b1, acc[5]);
    acc[6] = wmma_bf16(a1, b2, acc[6]);
    acc[7] = wmma_bf16(a1, b3, acc[7]);

    float* dst = covp + (size_t)z * C_DIM * C_DIM;
    const int r0 = (lane >> 4) * 8, cc = lane & 15;
    #pragma unroll
    for (int h = 0; h < 2; ++h)            // M halves: i0 and i0+16
        #pragma unroll
        for (int t4 = 0; t4 < 4; ++t4)
            #pragma unroll
            for (int r = 0; r < 8; ++r)
                dst[(size_t)(i0 + 16 * h + r0 + r) * C_DIM + j0 + 16 * t4 + cc] =
                    acc[4 * h + t4][r];
}

// ---- 4) reduce partials -> sigma (fp32), / (N-1) ---------------------------
__global__ void __launch_bounds__(256) k_cov_reduce(const float* __restrict__ covp,
                                                    float* __restrict__ sigma) {
    const size_t idx = (size_t)blockIdx.x * 256 + threadIdx.x;   // 1024 blocks
    float s = 0.f;
    #pragma unroll 4
    for (int z = 0; z < SPLITS; ++z)
        s += covp[(size_t)z * C_DIM * C_DIM + idx];
    sigma[idx] = s * (1.0f / (float)(N_ROWS - 1));
}

// ---- 5) trace of sigma, derived scalars ------------------------------------
__global__ void __launch_bounds__(512) k_trace(const float* __restrict__ sigma,
                                               float* __restrict__ scal) {
    __shared__ float red[512];
    const int t = threadIdx.x;
    red[t] = sigma[(size_t)t * C_DIM + t];
    __syncthreads();
    for (int s = 256; s > 0; s >>= 1) {
        if (t < s) red[t] += red[t + s];
        __syncthreads();
    }
    if (t == 0) {
        float tr = red[0];
        scal[0] = tr;
        scal[1] = 1.0f / tr;        // for sigma_n
        scal[2] = rsqrtf(tr);       // for wm = P / sqrt(tr)
    }
}

// ---- 6) sigma_n (bf16) and P = I -------------------------------------------
__global__ void __launch_bounds__(256) k_make_sigman_initP(const float* __restrict__ sigma,
                                                           const float* __restrict__ scal,
                                                           bf16* __restrict__ sigman,
                                                           bf16* __restrict__ P) {
    const size_t idx = (size_t)blockIdx.x * 256 + threadIdx.x;
    const int i = (int)(idx >> 9), j = (int)(idx & 511);
    const float inv = scal[1];
    sigman[idx] = (bf16)(sigma[idx] * inv);
    P[idx]      = (bf16)(i == j ? 1.0f : 0.0f);
}

// ---- 7) 512x512 bf16 WMMA GEMM; mode 1 fuses the NS update -----------------
// All operands symmetric -> B^T fetched as B row-major.
__global__ void __launch_bounds__(32) k_gemm512(const bf16* __restrict__ A,
                                                const bf16* __restrict__ BT,
                                                bf16* __restrict__ D,
                                                const bf16* __restrict__ Pprev,
                                                int mode) {
    const int lane = threadIdx.x;
    const int i0 = blockIdx.x * 16, j0 = blockIdx.y * 64;
    v8f acc[4] = {};
    for (int k = 0; k < C_DIM; k += 32) {
        v16bf a = load_frag_a(A, C_DIM, i0, k, lane);
        #pragma unroll
        for (int t4 = 0; t4 < 4; ++t4) {
            v16bf b = load_frag_bt(BT, C_DIM, j0 + 16 * t4, k, lane);
            acc[t4] = wmma_bf16(a, b, acc[t4]);
        }
    }
    const int r0 = (lane >> 4) * 8, cc = lane & 15;
    #pragma unroll
    for (int t4 = 0; t4 < 4; ++t4)
        #pragma unroll
        for (int r = 0; r < 8; ++r) {
            size_t idx = (size_t)(i0 + r0 + r) * C_DIM + j0 + 16 * t4 + cc;
            float v = acc[t4][r];
            if (mode) v = 1.5f * (float)Pprev[idx] - 0.5f * v;   // P <- 1.5P - 0.5 P3@Sn
            D[idx] = (bf16)v;
        }
}

// ---- 8) out = (Xn_bf @ P) * rsqrt(trace); 16x64 per wave, double-buffered --
__global__ void __launch_bounds__(32) k_final_gemm(const bf16* __restrict__ xnbf,
                                                   const bf16* __restrict__ W,
                                                   const float* __restrict__ scal,
                                                   float* __restrict__ out) {
    const int lane = threadIdx.x;
    const int m0 = blockIdx.x * 16;
    const int j0 = blockIdx.y * 64;
    v8f acc[4] = {};

    v16bf a  = load_frag_a(xnbf, C_DIM, m0, 0, lane);
    v16bf b0 = load_frag_bt(W, C_DIM, j0,      0, lane);
    v16bf b1 = load_frag_bt(W, C_DIM, j0 + 16, 0, lane);
    v16bf b2 = load_frag_bt(W, C_DIM, j0 + 32, 0, lane);
    v16bf b3 = load_frag_bt(W, C_DIM, j0 + 48, 0, lane);

    for (int k = 32; k < C_DIM; k += 32) {
        v16bf na  = load_frag_a(xnbf, C_DIM, m0, k, lane);
        v16bf nb0 = load_frag_bt(W, C_DIM, j0,      k, lane);
        v16bf nb1 = load_frag_bt(W, C_DIM, j0 + 16, k, lane);
        v16bf nb2 = load_frag_bt(W, C_DIM, j0 + 32, k, lane);
        v16bf nb3 = load_frag_bt(W, C_DIM, j0 + 48, k, lane);
        acc[0] = wmma_bf16(a, b0, acc[0]);
        acc[1] = wmma_bf16(a, b1, acc[1]);
        acc[2] = wmma_bf16(a, b2, acc[2]);
        acc[3] = wmma_bf16(a, b3, acc[3]);
        a = na; b0 = nb0; b1 = nb1; b2 = nb2; b3 = nb3;
    }
    acc[0] = wmma_bf16(a, b0, acc[0]);
    acc[1] = wmma_bf16(a, b1, acc[1]);
    acc[2] = wmma_bf16(a, b2, acc[2]);
    acc[3] = wmma_bf16(a, b3, acc[3]);

    const float s = scal[2];
    const int r0 = (lane >> 4) * 8, cc = lane & 15;
    #pragma unroll
    for (int t4 = 0; t4 < 4; ++t4)
        #pragma unroll
        for (int r = 0; r < 8; ++r)
            out[(size_t)(m0 + r0 + r) * C_DIM + j0 + 16 * t4 + cc] = acc[t4][r] * s;
}

// ---------------------------------------------------------------------------
extern "C" void kernel_launch(void* const* d_in, const int* in_sizes, int n_in,
                              void* d_out, int out_size, void* d_ws, size_t ws_size,
                              hipStream_t stream) {
    (void)in_sizes; (void)n_in; (void)out_size; (void)ws_size;
    const float* x = (const float*)d_in[0];
    float* out = (float*)d_out;

    char* ws = (char*)d_ws;
    size_t off = 0;
    float* meanp  = (float*)(ws + off); off += (size_t)256 * C_DIM * 4;          // 512 KB
    float* mean   = (float*)(ws + off); off += (size_t)C_DIM * 4;                // 2 KB
    off = (off + 255) & ~(size_t)255;
    float* scal   = (float*)(ws + off); off += 256;                              // scalars
    float* sigma  = (float*)(ws + off); off += (size_t)C_DIM * C_DIM * 4;        // 1 MB
    bf16*  sigman = (bf16*) (ws + off); off += (size_t)C_DIM * C_DIM * 2;        // 512 KB
    bf16*  P      = (bf16*) (ws + off); off += (size_t)C_DIM * C_DIM * 2;
    bf16*  P2     = (bf16*) (ws + off); off += (size_t)C_DIM * C_DIM * 2;
    bf16*  P3     = (bf16*) (ws + off); off += (size_t)C_DIM * C_DIM * 2;
    float* covp   = (float*)(ws + off); off += (size_t)SPLITS * C_DIM * C_DIM * 4; // 32 MB
    bf16*  xnbf   = (bf16*) (ws + off); off += (size_t)N_ROWS * C_DIM * 2;       // 64 MB
    bf16*  xnT    = (bf16*) (ws + off); off += (size_t)N_ROWS * C_DIM * 2;       // 64 MB

    k_mean_partial     <<<256, 512, 0, stream>>>(x, meanp);
    k_mean_final       <<<1,   512, 0, stream>>>(meanp, mean);
    k_center_transpose <<<dim3(N_ROWS / 64, C_DIM / 64), 256, 0, stream>>>(x, mean, xnbf, xnT);
    k_cov_partial      <<<dim3(C_DIM / 32, C_DIM / 64, SPLITS), 32, 0, stream>>>(xnT, covp);
    k_cov_reduce       <<<(C_DIM * C_DIM) / 256, 256, 0, stream>>>(covp, sigma);
    k_trace            <<<1, 512, 0, stream>>>(sigma, scal);
    k_make_sigman_initP<<<(C_DIM * C_DIM) / 256, 256, 0, stream>>>(sigma, scal, sigman, P);

    for (int it = 0; it < 5; ++it) {
        k_gemm512<<<dim3(C_DIM / 16, C_DIM / 64), 32, 0, stream>>>(P,  P,      P2, nullptr, 0);
        k_gemm512<<<dim3(C_DIM / 16, C_DIM / 64), 32, 0, stream>>>(P2, P,      P3, nullptr, 0);
        k_gemm512<<<dim3(C_DIM / 16, C_DIM / 64), 32, 0, stream>>>(P3, sigman, P,  P,       1);
    }

    k_final_gemm<<<dim3(N_ROWS / 16, C_DIM / 64), 32, 0, stream>>>(xnbf, P, scal, out);
}
